// MultiHeadSelfAttention_1649267442188
// MI455X (gfx1250) — matrix operand order repaired and checked by reading, not compiled, not run on hardware
//
#include <hip/hip_runtime.h>

// ---------------------------------------------------------------------------
// Multi-head self-attention for MI455X (gfx1250, wave32, WMMA bf16).
//  B=2, T=2048, C=1024, H=16, Dh=64.
//  Stage 0: fp32 -> bf16 conversion of X / W_qkv / W_out (once, outside loops)
//  Stage 1: QKV GEMM (32x64 tiles/wave, pure bf16 WMMA inner loop)
//  Stage 2: fused flash attention; 8 waves (8 query tiles) of one head per
//           workgroup cooperatively stage each 32-key K/V tile in LDS (8x
//           less K/V traffic), S^T = K*Q^T so softmax is per-lane.
//  Stage 3: output projection GEMM (bf16 WMMA, fp32 epilogue)
// ---------------------------------------------------------------------------

#define BATCH   2
#define SEQ     2048
#define DMODEL  1024
#define NHEADS  16
#define HDIM    64
#define MROWS   (BATCH * SEQ)        // 4096
#define NQKV    (3 * DMODEL)         // 3072

typedef __bf16 v16bf __attribute__((ext_vector_type(16)));
typedef __bf16 v8bf  __attribute__((ext_vector_type(8)));
typedef float  v8f   __attribute__((ext_vector_type(8)));

__device__ __forceinline__ v8f wmma_bf16(v16bf a, v16bf b, v8f c) {
    return __builtin_amdgcn_wmma_f32_16x16x32_bf16(
        false, a, false, b, (short)0, c, false, false);
}

// ---- WMMA fragment loaders from packed bf16 (ISA 7.12.2 layouts) ----------
// A 16x32: lane holds row M=lane&15; lanes<16 K={0..7,16..23}, lanes>=16
// K={8..15,24..31}.  Two 16-byte vector loads.
__device__ __forceinline__ v16bf load_a16(const __bf16* A,
                                          int lda, int row0, int k0, int lane) {
    const int m  = lane & 15;
    const int kb = (lane < 16) ? 0 : 8;
    const __bf16* p = A + (size_t)(row0 + m) * lda + k0 + kb;
    v8bf lo = *(const v8bf*)p;
    v8bf hi = *(const v8bf*)(p + 16);
    return __builtin_shufflevector(lo, hi, 0, 1, 2, 3, 4, 5, 6, 7,
                                   8, 9, 10, 11, 12, 13, 14, 15);
}
// B 32x16: lane holds column N=lane&15; lanes<16 K={0..7,16..23}, lanes>=16
// K={8..15,24..31}.
__device__ __forceinline__ v16bf load_b16(const __bf16* B,
                                          int ldb, int k0, int col0, int lane) {
    const int n  = lane & 15;
    const int kb = (lane < 16) ? 0 : 8;
    const __bf16* p = B + (size_t)(k0 + kb) * ldb + col0 + n;
    v16bf r;
#pragma unroll
    for (int s = 0; s < 8; ++s) {
        r[s]     = p[(size_t)s * ldb];
        r[s + 8] = p[(size_t)(s + 16) * ldb];
    }
    return r;
}

// ---------------------------------------------------------------------------
// Stage 0: fp32 -> bf16 bulk convert (8 elems/thread, b128 in / b128 out).
// ---------------------------------------------------------------------------
__global__ __launch_bounds__(256) void cvt_bf16_kernel(
    const float* __restrict__ src, __bf16* __restrict__ dst, int n) {
    const int idx = (blockIdx.x * blockDim.x + threadIdx.x) * 8;
    if (idx + 8 > n) return;
    const float4 a = *(const float4*)(src + idx);
    const float4 b = *(const float4*)(src + idx + 4);
    v8bf o;
    o[0] = (__bf16)a.x; o[1] = (__bf16)a.y; o[2] = (__bf16)a.z; o[3] = (__bf16)a.w;
    o[4] = (__bf16)b.x; o[5] = (__bf16)b.y; o[6] = (__bf16)b.z; o[7] = (__bf16)b.w;
    *(v8bf*)(dst + idx) = o;
}

// ---------------------------------------------------------------------------
// Stage 1: QKV = Xb(4096x1024) @ Wb(1024x3072) + b ; scatter bf16 Q/K/V in
// [B,H,T,Dh] layout (Q pre-scaled by 1/sqrt(Dh)).  32x64 tile per wave.
// ---------------------------------------------------------------------------
__global__ __launch_bounds__(256) void qkv_gemm_kernel(
    const __bf16* __restrict__ Xb, const __bf16* __restrict__ Wb,
    const float* __restrict__ bias,
    __bf16* __restrict__ Qb, __bf16* __restrict__ Kb, __bf16* __restrict__ Vb) {
    const int lane = threadIdx.x & 31;
    const int gw   = (blockIdx.x * blockDim.x + threadIdx.x) >> 5;
    const int ntile = NQKV / 64;                       // 48
    const int tm = gw / ntile, tn = gw % ntile;
    const int row0 = tm * 32, col0 = tn * 64;

    const v8f z = {0.f, 0.f, 0.f, 0.f, 0.f, 0.f, 0.f, 0.f};
    v8f acc[2][4] = {{z, z, z, z}, {z, z, z, z}};

    for (int k0 = 0; k0 < DMODEL; k0 += 32) {
        v16bf a0 = load_a16(Xb, DMODEL, row0,      k0, lane);
        v16bf a1 = load_a16(Xb, DMODEL, row0 + 16, k0, lane);
#pragma unroll
        for (int t = 0; t < 4; ++t) {
            v16bf b = load_b16(Wb, NQKV, k0, col0 + 16 * t, lane);
            acc[0][t] = wmma_bf16(a0, b, acc[0][t]);
            acc[1][t] = wmma_bf16(a1, b, acc[1][t]);
        }
    }

    const int mo = (lane < 16) ? 0 : 8;
    const int nl = lane & 15;
#pragma unroll
    for (int t = 0; t < 4; ++t) {
        const int n   = col0 + 16 * t + nl;
        const int sec = n >> 10;                       // 0=Q 1=K 2=V
        const int c   = n & (DMODEL - 1);
        const int h   = c >> 6, d = c & 63;
        __bf16* dst = (sec == 0) ? Qb : (sec == 1) ? Kb : Vb;
        const float scl = (sec == 0) ? 0.125f : 1.0f;  // fold 1/sqrt(Dh) into Q
        const float bv = bias[n];
#pragma unroll
        for (int s = 0; s < 2; ++s) {
#pragma unroll
            for (int i = 0; i < 8; ++i) {
                const int r = row0 + s * 16 + mo + i;  // 0..4095
                const int bi = r >> 11, tpos = r & (SEQ - 1);
                const size_t idx =
                    ((size_t)(bi * NHEADS + h) * SEQ + tpos) * HDIM + d;
                dst[idx] = (__bf16)((acc[s][t][i] + bv) * scl);
            }
        }
    }
}

// ---------------------------------------------------------------------------
// Stage 2: fused flash attention.  One block (8 waves) per (b,h, 128-query
// group); each wave owns one 16-query tile.  Each 32-key step: the block
// cooperatively copies the 32x64 K and V tiles into LDS (one b128 per thread
// per tile), then every wave computes St = K*Q^T with WMMA, per-lane online
// softmax, P restaged through per-wave LDS into the A layout, O += P*V.
// ---------------------------------------------------------------------------
__global__ __launch_bounds__(256) void attn_flash_kernel(
    const __bf16* __restrict__ Qb, const __bf16* __restrict__ Kb,
    const __bf16* __restrict__ Vb, __bf16* __restrict__ attnb) {
    __shared__ __align__(16) __bf16 kbuf[32 * HDIM];     // 4KB
    __shared__ __align__(16) __bf16 vbuf[32 * HDIM];     // 4KB
    __shared__ __align__(16) __bf16 pstage[8][16 * 32];  // 8KB

    const int tid  = threadIdx.x;
    const int lane = tid & 31;
    const int wv   = tid >> 5;
    const int bh     = blockIdx.x >> 4;                // 0..31
    const int qgroup = blockIdx.x & 15;                // 0..15
    const int q0 = (qgroup * 8 + wv) * 16;             // wave's query tile

    const size_t base = (size_t)bh * SEQ * HDIM;
    const __bf16* Qh = Qb + base;
    const __bf16* Kh = Kb + base;
    const __bf16* Vh = Vb + base;
    __bf16* pst = &pstage[wv][0];

    const int mo = (lane < 16) ? 0 : 8;
    const int nl = lane & 15;
    const int hh = lane >> 4;                          // half-wave id

    // Q^T B-fragments (hoisted strided loads, once per wave).
    // B(k=d, n=query): lane = query nl, element j = d (j&7)+8*hh+16*(j>>3).
    v16bf qt0, qt1;
#pragma unroll
    for (int j = 0; j < 16; ++j) {
        qt0[j] = Qh[(size_t)(q0 + nl) * HDIM + (j & 7) + 8 * hh + 16 * (j >> 3)];
        qt1[j] = Qh[(size_t)(q0 + nl) * HDIM + 32 + (j & 7) + 8 * hh + 16 * (j >> 3)];
    }

    const v8f z = {0.f, 0.f, 0.f, 0.f, 0.f, 0.f, 0.f, 0.f};
    v8f o[4] = {z, z, z, z};
    float mrun = -1e30f, lrun = 0.f;                   // per-lane (query=nl)

    for (int kb0 = 0; kb0 < SEQ; kb0 += 32) {
        // ---- cooperative K/V tile staging: 32x64 bf16 each, 1 b128/thread --
        __syncthreads();                               // previous tile consumed
        {
            const size_t g = (size_t)kb0 * HDIM + tid * 8;   // linear, rows contiguous
            *(v8bf*)&kbuf[tid * 8] = *(const v8bf*)(Kh + g);
            *(v8bf*)&vbuf[tid * 8] = *(const v8bf*)(Vh + g);
        }
        __syncthreads();

        // ---- St tiles from LDS: rows = keys, cols = queries ----
        v8f st0 = z, st1 = z;
        {
            v16bf ka = load_a16(kbuf, HDIM, 0, 0, lane);     // keys 0-15, d 0-31
            st0 = wmma_bf16(ka, qt0, st0);
            ka = load_a16(kbuf, HDIM, 0, 32, lane);          // d 32-63
            st0 = wmma_bf16(ka, qt1, st0);
            ka = load_a16(kbuf, HDIM, 16, 0, lane);          // keys 16-31
            st1 = wmma_bf16(ka, qt0, st1);
            ka = load_a16(kbuf, HDIM, 16, 32, lane);
            st1 = wmma_bf16(ka, qt1, st1);
        }

        // ---- per-lane online softmax (lane owns query nl in half hh) ----
        float bmax = -1e30f;
#pragma unroll
        for (int i = 0; i < 8; ++i) bmax = fmaxf(bmax, fmaxf(st0[i], st1[i]));
        bmax = fmaxf(bmax, __shfl_xor(bmax, 16, 32));
        const float nm = fmaxf(mrun, bmax);
        const float sc = __expf(mrun - nm);
        v8bf p0v, p1v;
        float rsum = 0.f;
#pragma unroll
        for (int i = 0; i < 8; ++i) {
            const float p0 = __expf(st0[i] - nm);
            const float p1 = __expf(st1[i] - nm);
            rsum += p0 + p1;
            p0v[i] = (__bf16)p0;
            p1v[i] = (__bf16)p1;
        }
        rsum += __shfl_xor(rsum, 16, 32);
        lrun = lrun * sc + rsum;
        mrun = nm;

        // rescale O: row i of O needs sc of query (i+mo), held in lane (i+mo)
#pragma unroll
        for (int i = 0; i < 8; ++i) {
            const float si = __shfl(sc, i + mo, 32);
            o[0][i] *= si; o[1][i] *= si; o[2][i] *= si; o[3][i] *= si;
        }

        // ---- restage P^T(D-layout) -> P(A-layout) via per-wave LDS ----
        // lane's keys in tile0 are k=i+8*hh, tile1 k=16+i+8*hh; query row nl.
        *(v8bf*)&pst[nl * 32 + 8 * hh]      = p0v;
        *(v8bf*)&pst[nl * 32 + 16 + 8 * hh] = p1v;
        asm volatile("s_wait_dscnt 0" ::: "memory");   // same-wave cross-lane RAW

        v16bf pa = load_a16(pst, 32, 0, 0, lane);      // P 16q x 32k
#pragma unroll
        for (int t = 0; t < 4; ++t) {
            v16bf bv = load_b16(vbuf, HDIM, 0, t * 16, lane);
            o[t] = wmma_bf16(pa, bv, o[t]);
        }
    }

    // ---- finalize: divide by row sum, store bf16 attn in [B,T,C] ----
    const float linv = 1.f / lrun;
    const int bi = bh >> 4, h = bh & 15;
#pragma unroll
    for (int i = 0; i < 8; ++i) {
        const float lv = __shfl(linv, i + mo, 32);
        const int r = q0 + mo + i;
        const size_t rowbase = ((size_t)bi * SEQ + r) * DMODEL + h * HDIM;
#pragma unroll
        for (int t = 0; t < 4; ++t)
            attnb[rowbase + t * 16 + nl] = (__bf16)(o[t][i] * lv);
    }
}

// ---------------------------------------------------------------------------
// Stage 3: out = attnb(4096x1024) @ Woutb(1024x1024) + b_out (fp32 out).
// 32x64 tile per wave.
// ---------------------------------------------------------------------------
__global__ __launch_bounds__(256) void oproj_gemm_kernel(
    const __bf16* __restrict__ A, const __bf16* __restrict__ W,
    const float* __restrict__ bias, float* __restrict__ out) {
    const int lane = threadIdx.x & 31;
    const int gw   = (blockIdx.x * blockDim.x + threadIdx.x) >> 5;
    const int ntile = DMODEL / 64;                     // 16
    const int tm = gw / ntile, tn = gw % ntile;
    const int row0 = tm * 32, col0 = tn * 64;

    const v8f z = {0.f, 0.f, 0.f, 0.f, 0.f, 0.f, 0.f, 0.f};
    v8f acc[2][4] = {{z, z, z, z}, {z, z, z, z}};

    for (int k0 = 0; k0 < DMODEL; k0 += 32) {
        v16bf a0 = load_a16(A, DMODEL, row0,      k0, lane);
        v16bf a1 = load_a16(A, DMODEL, row0 + 16, k0, lane);
#pragma unroll
        for (int t = 0; t < 4; ++t) {
            v16bf b = load_b16(W, DMODEL, k0, col0 + 16 * t, lane);
            acc[0][t] = wmma_bf16(a0, b, acc[0][t]);
            acc[1][t] = wmma_bf16(a1, b, acc[1][t]);
        }
    }

    const int mo = (lane < 16) ? 0 : 8;
    const int nl = lane & 15;
#pragma unroll
    for (int t = 0; t < 4; ++t) {
        const int n = col0 + 16 * t + nl;
        const float bv = bias[n];
#pragma unroll
        for (int s = 0; s < 2; ++s) {
#pragma unroll
            for (int i = 0; i < 8; ++i) {
                const int r = row0 + s * 16 + mo + i;
                out[(size_t)r * DMODEL + n] = acc[s][t][i] + bv;
            }
        }
    }
}

// ---------------------------------------------------------------------------
extern "C" void kernel_launch(void* const* d_in, const int* in_sizes, int n_in,
                              void* d_out, int out_size, void* d_ws,
                              size_t ws_size, hipStream_t stream) {
    (void)in_sizes; (void)n_in; (void)out_size; (void)ws_size;
    const float* x     = (const float*)d_in[0];   // [2,2048,1024]
    const float* W_qkv = (const float*)d_in[1];   // [1024,3072]
    const float* b_qkv = (const float*)d_in[2];   // [3072]
    const float* W_out = (const float*)d_in[3];   // [1024,1024]
    const float* b_out = (const float*)d_in[4];   // [1024]
    float* out = (float*)d_out;                   // [2,2048,1024]

    // Workspace layout (all bf16): Xb | Wqkvb | Woutb | Q | K | V | attnb
    const size_t nX   = (size_t)MROWS * DMODEL;       // 4,194,304
    const size_t nWq  = (size_t)DMODEL * NQKV;        // 3,145,728
    const size_t nWo  = (size_t)DMODEL * DMODEL;      // 1,048,576
    const size_t nQKV = (size_t)BATCH * NHEADS * SEQ * HDIM;  // 4,194,304
    __bf16* Xb    = (__bf16*)d_ws;
    __bf16* Wqkvb = Xb + nX;
    __bf16* Woutb = Wqkvb + nWq;
    __bf16* Qb    = Woutb + nWo;
    __bf16* Kb    = Qb + nQKV;
    __bf16* Vb    = Kb + nQKV;
    __bf16* attnb = Vb + nQKV;                        // ~48MB total

    // Stage 0: bulk fp32->bf16 (8 elems/thread)
    cvt_bf16_kernel<<<(int)(nX  / 8 / 256), 256, 0, stream>>>(x,     Xb,    (int)nX);
    cvt_bf16_kernel<<<(int)(nWq / 8 / 256), 256, 0, stream>>>(W_qkv, Wqkvb, (int)nWq);
    cvt_bf16_kernel<<<(int)(nWo / 8 / 256), 256, 0, stream>>>(W_out, Woutb, (int)nWo);

    // Stage 1: (4096/32)*(3072/64) = 6144 waves, 8 waves/block.
    qkv_gemm_kernel<<<6144 / 8, 256, 0, stream>>>(Xb, Wqkvb, b_qkv, Qb, Kb, Vb);
    // Stage 2: one block per (b,h, 128-query group): 32*16 = 512 blocks.
    attn_flash_kernel<<<512, 256, 0, stream>>>(Qb, Kb, Vb, attnb);
    // Stage 3: (4096/32)*(1024/64) = 2048 waves.
    oproj_gemm_kernel<<<2048 / 8, 256, 0, stream>>>(attnb, Woutb, b_out, out);
}
